// RNN_76424648065820
// MI455X (gfx1250) — matrix operand order repaired and checked by reading, not compiled, not run on hardware
//
#include <hip/hip_runtime.h>
#include <hip/hip_bf16.h>

// ---------------------------------------------------------------------------
// GRU (2 layers, B=64, T=512, H=600) for gfx1250.
//   Phase A (per layer): gi = x @ W_ih^T + b_ih  -> big WMMA f16 GEMM, f32 acc
//   Phase B (per layer): persistent kernel, 512 sequential steps, WMMA for
//     gh = h @ W_hh^T (weights resident in LDS), gate math in registers,
//     one grid barrier (L2 atomics + s_sleep) per step, double-buffered h16.
// Workspace use: ~293 MB (gi buffer dominates).
// ---------------------------------------------------------------------------

typedef __attribute__((ext_vector_type(16))) _Float16 v16h;
typedef __attribute__((ext_vector_type(8)))  _Float16 v8h;
typedef __attribute__((ext_vector_type(8)))  float    v8f;

#define B_    64
#define T_    512
#define H_    600
#define NP    608          // H padded to multiple of 16 (and 32 for K use)
#define GIP   1824         // 3*NP, padded 3H
#define D0    300
#define KP0   320          // layer0 input dim padded to mult of 32
#define M_    32768        // B*T
#define NTILE 114          // GIP/16
#define NWG_SEQ 38         // NP/16 workgroups in the persistent kernel

// Build a v16h WMMA fragment from two 16-byte chunks (works for global & LDS).
__device__ __forceinline__ v16h ldfrag(const _Float16* p, int o0, int o1) {
  v8h lo = *(const v8h*)(p + o0);
  v8h hi = *(const v8h*)(p + o1);
  return __builtin_shufflevector(lo, hi, 0,1,2,3,4,5,6,7,8,9,10,11,12,13,14,15);
}
// A-matrix 16x32 f16 layout: lanes0-15 K=0..7 & 16..23, lanes16-31 K=8..15 & 24..31
__device__ __forceinline__ v16h ldA(const _Float16* row, int k0, int laneHi) {
  int alo = laneHi ? 8 : 0;
  return ldfrag(row, k0 + alo, k0 + 16 + alo);
}
// B-matrix 32x16 f16 layout: lane holds column n=lane%16; lanes0-15 K=0..7 & 16..23, lanes16-31 K=8..15 & 24..31
__device__ __forceinline__ v16h ldB(const _Float16* row, int k0, int laneHi) {
  int sel = laneHi ? 8 : 0;
  return ldfrag(row, k0 + sel, k0 + 16 + sel);
}

__device__ __forceinline__ float sigmoid_(float x) { return 1.f / (1.f + __expf(-x)); }
__device__ __forceinline__ float tanh_(float x)    { return 1.f - 2.f / (__expf(2.f * x) + 1.f); }

// --------------------------- conversion kernels ----------------------------
__global__ __launch_bounds__(256) void conv_x0(const float* __restrict__ x,
                                               _Float16* __restrict__ x16) {
  long i = (long)blockIdx.x * 256 + threadIdx.x;
  if (i >= (long)M_ * KP0) return;
  long bt = i / KP0; int k = (int)(i % KP0);
  x16[i] = (k < D0) ? (_Float16)x[bt * D0 + k] : (_Float16)0.f;
}

// src [3H][Kin] f32 -> dst [3*NP][KP] f16, zero padded in both dims (per-gate rows)
__global__ __launch_bounds__(256) void conv_w(const float* __restrict__ src,
                                              _Float16* __restrict__ dst,
                                              int Kin, int KP) {
  long i = (long)blockIdx.x * 256 + threadIdx.x;
  if (i >= (long)GIP * KP) return;
  int r = (int)(i / KP); int k = (int)(i % KP);
  int g = r / NP; int j = r - g * NP;
  dst[i] = (j < H_ && k < Kin) ? (_Float16)src[((long)(g * H_ + j)) * Kin + k]
                               : (_Float16)0.f;
}

__global__ __launch_bounds__(256) void init_h(const float* __restrict__ h0,
                                              _Float16* __restrict__ h16) {
  int i = blockIdx.x * 256 + threadIdx.x;
  if (i >= B_ * NP) return;
  int b = i / NP, j = i - b * NP;
  h16[i] = (j < H_) ? (_Float16)h0[b * H_ + j] : (_Float16)0.f;
}

// --------------------------- gi GEMM (parallel) ----------------------------
// gi[M][GIP] = X[M][KP] @ W[GIP][KP]^T + b_ih.  Wave computes 4 M-tiles x 1 N-tile.
__global__ __launch_bounds__(128) void gi_gemm(const _Float16* __restrict__ X,
                                               const _Float16* __restrict__ W,
                                               const float* __restrict__ bias,
                                               float* __restrict__ gi, int KP) {
  int wave = threadIdx.x >> 5, lane = threadIdx.x & 31;
  int laneHi = lane >> 4, nl = lane & 15;
  int waveId = blockIdx.x * 4 + wave;
  int mg = waveId / NTILE;          // 0..511 (group of 4 M-tiles)
  int nt = waveId % NTILE;          // 0..113
  int n = nt * 16 + nl;
  const _Float16* wrow = W + (long)n * KP;
  const _Float16* ar[4];
#pragma unroll
  for (int i = 0; i < 4; ++i) ar[i] = X + (long)(mg * 64 + i * 16 + nl) * KP;

  v8f acc[4] = {};
  for (int k0 = 0; k0 < KP; k0 += 32) {
    v16h bm = ldB(wrow, k0, laneHi);
#pragma unroll
    for (int i = 0; i < 4; ++i) {
      v16h am = ldA(ar[i], k0, laneHi);
      acc[i] = __builtin_amdgcn_wmma_f32_16x16x32_f16(false, am, false, bm,
                                                      (short)0, acc[i], false, false);
    }
  }
  int g = n / NP, j = n - g * NP;
  float bv = (j < H_) ? bias[g * H_ + j] : 0.f;
#pragma unroll
  for (int i = 0; i < 4; ++i)
#pragma unroll
    for (int v = 0; v < 8; ++v) {
      int row = mg * 64 + i * 16 + v + laneHi * 8;
      gi[(long)row * GIP + n] = acc[i][v] + bv;
    }
}

// ----------------------- persistent recurrent kernel -----------------------
__device__ __forceinline__ void grid_barrier(int* ctr) {
  __threadfence();
  __syncthreads();
  if (threadIdx.x == 0) {
    __hip_atomic_fetch_add(ctr, 1, __ATOMIC_RELEASE, __HIP_MEMORY_SCOPE_AGENT);
    while (__hip_atomic_load(ctr, __ATOMIC_ACQUIRE, __HIP_MEMORY_SCOPE_AGENT) < NWG_SEQ)
      __builtin_amdgcn_s_sleep(1);
  }
  __syncthreads();
}

__global__ __launch_bounds__(128)
void gru_seq(const _Float16* __restrict__ Whh16,   // [GIP][NP] f16, per-gate padded
             const float* __restrict__ bhh,        // [3H]
             const float* __restrict__ h0,         // [B][H] initial hidden
             const float* __restrict__ gi,         // [M][GIP]
             _Float16* __restrict__ h16,           // [2][B*NP] double buffer
             _Float16* __restrict__ xout16,        // layer0: [M][NP] (else unused)
             float* __restrict__ seqout,           // layer1: [B,T,H] (else unused)
             float* __restrict__ hout,             // [B][H] final hidden
             int* __restrict__ sync, int layer) {
  __shared__ _Float16 ldsW[3 * 16 * NP];           // 58368 B: this WG's W_hh slice
  const int wave = threadIdx.x >> 5, lane = threadIdx.x & 31;
  const int laneHi = lane >> 4, nl = lane & 15, hi8 = laneHi * 8;
  const int j0 = blockIdx.x * 16;
  const int m0 = wave * 16;
  const int j = j0 + nl;
  const bool jok = (j < H_);

  // Stage W_hh slice: rows {g*NP + j0 + i} -> ldsW[(g*16+i)*NP + k]  (once)
  for (int u = threadIdx.x; u < 3 * 16 * NP / 2; u += 128) {
    int h2 = u * 2;
    int k = h2 % NP, ri = h2 / NP;
    int g = ri >> 4, i = ri & 15;
    *(unsigned int*)&ldsW[(long)ri * NP + k] =
        *(const unsigned int*)(Whh16 + (long)(g * NP + j0 + i) * NP + k);
  }
  __syncthreads();

  const float bh0 = jok ? bhh[0 * H_ + j] : 0.f;
  const float bh1 = jok ? bhh[1 * H_ + j] : 0.f;
  const float bh2 = jok ? bhh[2 * H_ + j] : 0.f;

  // h tile resident in registers, WMMA C-layout: element v -> (b=m0+v+hi8, j)
  v8f hreg = {};
#pragma unroll
  for (int v = 0; v < 8; ++v)
    hreg[v] = jok ? h0[(m0 + v + hi8) * H_ + j] : 0.f;

  for (int t = 0; t < T_; ++t) {
    const _Float16* hc = h16 + (long)(t & 1) * (B_ * NP);
    _Float16* hn = h16 + (long)((t + 1) & 1) * (B_ * NP);

    // gh = h @ W_hh^T for the 3 gate tiles of this (m-tile, j-slice)
    v8f acc[3] = {};
    const _Float16* arow = hc + (long)(m0 + nl) * NP;
    for (int k0 = 0; k0 < NP; k0 += 32) {
      v16h am = ldA(arow, k0, laneHi);
#pragma unroll
      for (int g = 0; g < 3; ++g) {
        v16h bm = ldB(&ldsW[(long)(g * 16 + nl) * NP], k0, laneHi);
        acc[g] = __builtin_amdgcn_wmma_f32_16x16x32_f16(false, am, false, bm,
                                                        (short)0, acc[g], false, false);
      }
    }

    // gates + state update (local to this wave)
#pragma unroll
    for (int v = 0; v < 8; ++v) {
      int b = m0 + v + hi8;
      long bt = (long)b * T_ + t;
      const float* gr = gi + bt * GIP;
      float r = sigmoid_(gr[j] + acc[0][v] + bh0);
      float z = sigmoid_(gr[NP + j] + acc[1][v] + bh1);
      float nn = tanh_(gr[2 * NP + j] + r * (acc[2][v] + bh2));
      float hv = (1.f - z) * nn + z * hreg[v];
      hreg[v] = hv;
      if (jok) {
        hn[(long)b * NP + j] = (_Float16)hv;
        if (layer == 0) xout16[bt * NP + j] = (_Float16)hv;
        else            seqout[bt * H_ + j] = hv;
        if (t == T_ - 1) hout[b * H_ + j] = hv;
      }
    }
    grid_barrier(sync + t);
  }
}

// ------------------------------- launcher ----------------------------------
extern "C" void kernel_launch(void* const* d_in, const int* in_sizes, int n_in,
                              void* d_out, int out_size, void* d_ws, size_t ws_size,
                              hipStream_t stream) {
  (void)in_sizes; (void)n_in; (void)out_size; (void)ws_size;
  const float* x0   = (const float*)d_in[0];
  const float* hid  = (const float*)d_in[1];
  const float* Wih0 = (const float*)d_in[2];
  const float* Whh0 = (const float*)d_in[3];
  const float* bih0 = (const float*)d_in[4];
  const float* bhh0 = (const float*)d_in[5];
  const float* Wih1 = (const float*)d_in[6];
  const float* Whh1 = (const float*)d_in[7];
  const float* bih1 = (const float*)d_in[8];
  const float* bhh1 = (const float*)d_in[9];

  float* ret    = (float*)d_out;                       // [B,T,H]
  float* hidout = ret + (long)B_ * T_ * H_;            // [2,B,H]

  char* ws = (char*)d_ws;
  size_t off = 0;
  auto alloc = [&](size_t bytes) {
    char* p = ws + off; off = (off + bytes + 255) & ~(size_t)255; return p;
  };
  float*    gi    = (float*)   alloc((size_t)M_ * GIP * 4);
  _Float16* x16   = (_Float16*)alloc((size_t)M_ * KP0 * 2);
  _Float16* x116  = (_Float16*)alloc((size_t)M_ * NP * 2);
  _Float16* wih0h = (_Float16*)alloc((size_t)GIP * KP0 * 2);
  _Float16* whh0h = (_Float16*)alloc((size_t)GIP * NP * 2);
  _Float16* wih1h = (_Float16*)alloc((size_t)GIP * NP * 2);
  _Float16* whh1h = (_Float16*)alloc((size_t)GIP * NP * 2);
  _Float16* h16   = (_Float16*)alloc((size_t)2 * B_ * NP * 2);
  int*      sync  = (int*)     alloc((size_t)T_ * 4);

  (void)hipMemsetAsync(x116, 0, (size_t)M_ * NP * 2, stream);
  (void)hipMemsetAsync(sync, 0, (size_t)T_ * 4, stream);

  // f16 conversions (zero-padded)
  conv_x0<<<(int)(((long)M_ * KP0 + 255) / 256), 256, 0, stream>>>(x0, x16);
  conv_w <<<(int)(((long)GIP * KP0 + 255) / 256), 256, 0, stream>>>(Wih0, wih0h, D0, KP0);
  conv_w <<<(int)(((long)GIP * NP  + 255) / 256), 256, 0, stream>>>(Whh0, whh0h, H_, NP);
  conv_w <<<(int)(((long)GIP * NP  + 255) / 256), 256, 0, stream>>>(Wih1, wih1h, H_, NP);
  conv_w <<<(int)(((long)GIP * NP  + 255) / 256), 256, 0, stream>>>(Whh1, whh1h, H_, NP);

  const int gemm_blocks = 512 * NTILE / 4;  // 4 waves/block, 1 (mgroup,ntile) per wave

  // ----- layer 0 -----
  (void)hipMemsetAsync(h16, 0, (size_t)2 * B_ * NP * 2, stream);
  init_h<<<(B_ * NP + 255) / 256, 256, 0, stream>>>(hid, h16);
  gi_gemm<<<gemm_blocks, 128, 0, stream>>>(x16, wih0h, bih0, gi, KP0);
  gru_seq<<<NWG_SEQ, 128, 0, stream>>>(whh0h, bhh0, hid, gi, h16,
                                       x116, nullptr, hidout, sync, 0);

  // ----- layer 1 -----
  (void)hipMemsetAsync(sync, 0, (size_t)T_ * 4, stream);
  (void)hipMemsetAsync(h16, 0, (size_t)2 * B_ * NP * 2, stream);
  init_h<<<(B_ * NP + 255) / 256, 256, 0, stream>>>(hid + (long)B_ * H_, h16);
  gi_gemm<<<gemm_blocks, 128, 0, stream>>>(x116, wih1h, bih1, gi, NP);
  gru_seq<<<NWG_SEQ, 128, 0, stream>>>(whh1h, bhh1, hid + (long)B_ * H_, gi, h16,
                                       nullptr, ret, hidout + (long)B_ * H_, sync, 1);
}
